// TransformerPointerNet_64725157150941
// MI455X (gfx1250) — matrix operand order repaired and checked by reading, not compiled, not run on hardware
//
#include <hip/hip_runtime.h>

// ---------------- problem constants ----------------
#define S_  128
#define N_  256
#define D_  256
#define H_  8
#define DH_ 32
#define FF_ 1024
#define NL_ 3
#define T_  (S_ * N_)   // 32768 tokens

// ---------------- WMMA types ----------------
typedef __attribute__((ext_vector_type(16))) __bf16 v16bf;
typedef __attribute__((ext_vector_type(8)))  float  v8f;

union FragBF { v16bf v; unsigned short u[16]; };

struct alignas(8) US4 { unsigned short x, y, z, w; };

__device__ __forceinline__ unsigned short f2bfu(float f) {
    unsigned u = __float_as_uint(f);
    unsigned r = u + 0x7FFFu + ((u >> 16) & 1u);   // round to nearest even
    return (unsigned short)(r >> 16);
}

__device__ __forceinline__ v8f wmma_bf16(const FragBF& a, const FragBF& b, v8f c) {
    return __builtin_amdgcn_wmma_f32_16x16x32_bf16(false, a.v, false, b.v,
                                                   (short)0, c, false, false);
}

// Inverse of the A-fragment K mapping: k in [0,32) -> (hi, e).
// Any 4-aligned k chunk keeps hi fixed and e consecutive (enables b64 stores).
__device__ __forceinline__ void inv_a(int k, int& hi, int& e) {
    hi = k >> 4;
    e = k & 15;
}

// 32-byte fragment load from LDS (lane's 16 bf16 contiguous -> 2x ds_load_b128)
__device__ __forceinline__ FragBF ld_frag(const unsigned short* p) {
    FragBF f;
    f.v = *(const v16bf*)p;
    return f;
}

// ============================================================
// bf16 WMMA GEMM:  C[M x Nc] = act(alpha * A @ Wp + bias)
// A fp32 row-major (M x K); Wp bf16 in pre-permuted fragment order:
//   Wp[((kp*Nc + c)*2 + hi)*16 + e] = bf16(W[kp*32 + hi*16 + e][c])
// block = 256 threads (8 waves); block tile = 128 rows x 64 cols
// grid = (Nc/64, M/128)
// ============================================================
__global__ void __launch_bounds__(256)
gemm_wmma(const float* __restrict__ A, const unsigned short* __restrict__ Wp,
          const float* __restrict__ bias, float* __restrict__ C,
          int M, int K, int Nc, float alpha, int relu)
{
    // As: [band(8)][hi(2)][m(16)][e(16)]  (= (wave*32+lane)*16 per-lane base)
    __shared__ unsigned short As[128 * 32];       // 8 KB
    // Ws: [ct(4)][hi*16+m (32 lanes)][e(16)]
    __shared__ unsigned short Ws[4 * 32 * 16];    // 4 KB

    const int tid  = threadIdx.x;
    const int wave = tid >> 5;
    const int lane = tid & 31;
    const int rowBase = blockIdx.y * 128;
    const int colBase = blockIdx.x * 64;
    const int m  = lane & 15;
    const int hi = lane >> 4;

    v8f acc0 = {}, acc1 = {}, acc2 = {}, acc3 = {};

    for (int kt = 0; kt < K; kt += 32) {
        // ---- stage A tile: 128x32 fp32 -> bf16 fragment order ----
        // each thread: float4 load + one 8B LDS store; 4 iterations
        #pragma unroll
        for (int it = 0; it < 4; ++it) {
            int i = tid + it * 256;                // 0..1023 chunk id
            int r = i >> 3, c = (i & 7) * 4;       // row, 4-aligned k
            const float4 va = *(const float4*)&A[(size_t)(rowBase + r) * K + (kt + c)];
            int ahi, ae; inv_a(c, ahi, ae);
            US4 w4 = { f2bfu(va.x), f2bfu(va.y), f2bfu(va.z), f2bfu(va.w) };
            *(US4*)&As[(((r >> 4) * 2 + ahi) * 16 + (r & 15)) * 16 + ae] = w4;
            if (kt + 32 < K)
                __builtin_prefetch(&A[(size_t)(rowBase + r) * K + (kt + 32 + c)], 0, 1);
        }
        // ---- stage W panel: pre-permuted, pure 16B copies (1 per thread) ----
        {
            int c   = tid >> 2;                    // col within block (0..63)
            int sub = tid & 3;                     // 8-ushort chunk within col
            int whi = sub >> 1, e8 = (sub & 1) * 8;
            int kp  = kt >> 5;
            const uint4* src = (const uint4*)
                (Wp + (((size_t)kp * Nc + colBase + c) * 2 + whi) * 16 + e8);
            uint4* dst = (uint4*)
                (Ws + (c >> 4) * 512 + whi * 256 + (c & 15) * 16 + e8);
            *dst = *src;
        }
        __syncthreads();

        FragBF a  = ld_frag(As + (wave * 32 + lane) * 16);
        FragBF b0 = ld_frag(Ws + 0 * 512 + lane * 16);
        FragBF b1 = ld_frag(Ws + 1 * 512 + lane * 16);
        FragBF b2 = ld_frag(Ws + 2 * 512 + lane * 16);
        FragBF b3 = ld_frag(Ws + 3 * 512 + lane * 16);
        acc0 = wmma_bf16(a, b0, acc0);
        acc1 = wmma_bf16(a, b1, acc1);
        acc2 = wmma_bf16(a, b2, acc2);
        acc3 = wmma_bf16(a, b3, acc3);
        __syncthreads();
    }

    // epilogue: 4 col tiles
    v8f accs[4] = {acc0, acc1, acc2, acc3};
    #pragma unroll
    for (int ct = 0; ct < 4; ++ct) {
        int col = colBase + ct * 16 + m;
        float bv = bias ? bias[col] : 0.0f;
        #pragma unroll
        for (int r = 0; r < 8; ++r) {
            int row = rowBase + wave * 16 + r + hi * 8;
            float v = accs[ct][r] * alpha + bv;
            if (relu) v = fmaxf(v, 0.0f);
            C[(size_t)row * Nc + col] = v;
        }
    }
}

// ============================================================
// Fused encoder self-attention, one block per (n, head).
// qkv layout: [(s*N + n)*3D + {0,D,2D} + h*DH + d]
// ============================================================
#define SC_STRIDE 129
#define PS_STRIDE 144

__global__ void __launch_bounds__(256)
enc_attn(const float* __restrict__ qkv, const int* __restrict__ mask,
         float* __restrict__ o)
{
    extern __shared__ char smem[];
    unsigned short* Qs = (unsigned short*)smem;                 // 128*32 (A-perm)
    unsigned short* Ks = Qs + 128 * 32;                         // 128*32 (natural [s][d])
    unsigned short* Vs = Ks + 128 * 32;                         // 128*32 (B-perm)
    float*          Sc = (float*)(Vs + 128 * 32);               // 128*SC_STRIDE f32
    unsigned short* Ps = (unsigned short*)(Sc + 128 * SC_STRIDE); // 128*PS_STRIDE bf16

    const int n  = blockIdx.x / H_;
    const int hh = blockIdx.x % H_;
    const int tid  = threadIdx.x;
    const int wave = tid >> 5;
    const int lane = tid & 31;
    const int m  = lane & 15;
    const int hi = lane >> 4;

    // load + convert Q,K,V (128 x 32 each); float4 loads, vectorized stores
    #pragma unroll
    for (int it = 0; it < 4; ++it) {
        int i = tid + it * 256;                    // 0..1023
        int s = i >> 3, d4 = (i & 7) * 4;
        size_t base = ((size_t)s * N_ + n) * (3 * D_) + hh * DH_ + d4;
        float4 q = *(const float4*)&qkv[base];
        float4 k = *(const float4*)&qkv[base + D_];
        float4 v = *(const float4*)&qkv[base + 2 * D_];
        // Q: A-fragment order (contiguous 4)
        int ahi, ae; inv_a(d4, ahi, ae);
        US4 q4 = { f2bfu(q.x), f2bfu(q.y), f2bfu(q.z), f2bfu(q.w) };
        *(US4*)&Qs[(((s >> 4) * 2 + ahi) * 16 + (s & 15)) * 16 + ae] = q4;
        // K: natural [s][d] (contiguous 4)
        US4 k4 = { f2bfu(k.x), f2bfu(k.y), f2bfu(k.z), f2bfu(k.w) };
        *(US4*)&Ks[s * 32 + d4] = k4;
        // V: B-fragment order (stride-16 scatter in d)
        int vkt = s >> 5, vrem = s & 31, vhi = vrem >> 4, ve = vrem & 15;
        unsigned short* vd = &Vs[((vkt * 2 + vhi) * 32 + d4) * 16 + ve];
        vd[0]  = f2bfu(v.x);
        vd[16] = f2bfu(v.y);
        vd[32] = f2bfu(v.z);
        vd[48] = f2bfu(v.w);
    }
    __syncthreads();

    // scores = Q @ K^T / sqrt(DH); wave w -> row tile w, loop col tiles
    const float scale = 0.17677669529663687f;  // 1/sqrt(32)
    {
        FragBF a = ld_frag(Qs + (wave * 32 + lane) * 16);
        for (int jt = 0; jt < 8; ++jt) {
            FragBF b = ld_frag(Ks + (jt * 16 + m) * 32 + hi * 16);
            v8f acc = {};
            acc = wmma_bf16(a, b, acc);
            #pragma unroll
            for (int r = 0; r < 8; ++r) {
                int rrow = wave * 16 + r + hi * 8;
                int ccol = jt * 16 + m;
                float v = acc[r] * scale;
                if (mask[n * S_ + ccol] == 0) v = -1e9f;
                Sc[rrow * SC_STRIDE + ccol] = v;
            }
        }
    }
    __syncthreads();

    // row softmax (threads 0..127), write bf16 P in A-fragment order
    if (tid < 128) {
        float mx = -3.4e38f;
        for (int c = 0; c < 128; ++c) mx = fmaxf(mx, Sc[tid * SC_STRIDE + c]);
        float sum = 0.0f;
        for (int c = 0; c < 128; ++c) {
            float e = __expf(Sc[tid * SC_STRIDE + c] - mx);
            sum += e;
            Sc[tid * SC_STRIDE + c] = e;
        }
        float inv = 1.0f / sum;
        for (int c = 0; c < 128; ++c) {
            int kt = c >> 5, c32 = c & 31, phi, pe;
            inv_a(c32, phi, pe);
            Ps[tid * PS_STRIDE + (kt * 2 + phi) * 16 + pe] =
                f2bfu(Sc[tid * SC_STRIDE + c] * inv);
        }
    }
    __syncthreads();

    // O = P(128x128) @ V(128x32): wave w -> row tile w, 2 col tiles, K=128
    for (int jt = 0; jt < 2; ++jt) {
        v8f acc = {};
        #pragma unroll
        for (int kt = 0; kt < 4; ++kt) {
            FragBF a = ld_frag(Ps + (wave * 16 + m) * PS_STRIDE + (kt * 2 + hi) * 16);
            FragBF b = ld_frag(Vs + ((kt * 2 + hi) * 32 + jt * 16 + m) * 16);
            acc = wmma_bf16(a, b, acc);
        }
        #pragma unroll
        for (int r = 0; r < 8; ++r) {
            int s = wave * 16 + r + hi * 8;
            int d = jt * 16 + m;
            o[((size_t)s * N_ + n) * D_ + hh * DH_ + d] = acc[r];
        }
    }
}

// ============================================================
// h = LayerNorm(h + delta) * s + b   (one block per token, 256 thr = D)
// ============================================================
__global__ void __launch_bounds__(256)
add_ln(float* __restrict__ h, const float* __restrict__ delta,
       const float* __restrict__ sc, const float* __restrict__ bi)
{
    __shared__ float red[256];
    const int t = blockIdx.x;
    const int d = threadIdx.x;
    const size_t idx = (size_t)t * D_ + d;
    float x = h[idx] + delta[idx];

    red[d] = x; __syncthreads();
    for (int off = 128; off > 0; off >>= 1) {
        if (d < off) red[d] += red[d + off];
        __syncthreads();
    }
    float mu = red[0] * (1.0f / D_);
    __syncthreads();
    float xm = x - mu;
    red[d] = xm * xm; __syncthreads();
    for (int off = 128; off > 0; off >>= 1) {
        if (d < off) red[d] += red[d + off];
        __syncthreads();
    }
    float var = red[0] * (1.0f / D_);
    h[idx] = xm * rsqrtf(var + 1e-5f) * sc[d] + bi[d];
}

// graph[n,d] = mean_s h[s,n,d]
__global__ void graph_mean(const float* __restrict__ h, float* __restrict__ g)
{
    int i = blockIdx.x * 256 + threadIdx.x;        // i in [0, N*D)
    int n = i >> 8, d = i & 255;
    float sum = 0.0f;
    for (int s = 0; s < S_; ++s) sum += h[((size_t)s * N_ + n) * D_ + d];
    g[i] = sum * (1.0f / S_);
}

// Convert fp32 weight (K x Nc) to bf16 in WMMA-B fragment order:
//   dst[((kp*Nc + c)*2 + hi)*16 + e] = bf16(src[(kp*32 + hi*16 + e)*Nc + c])
__global__ void f32_to_bf16_wperm(const float* __restrict__ src,
                                  unsigned short* __restrict__ dst, int K, int Nc)
{
    int i = blockIdx.x * 256 + threadIdx.x;
    if (i >= K * Nc) return;
    int k = i / Nc, c = i % Nc;
    int kp = k >> 5, k32 = k & 31, hi = k32 >> 4, e = k32 & 15;
    dst[(((size_t)kp * Nc + c) * 2 + hi) * 16 + e] = f2bfu(src[i]);
}

// ---------------- decode-side kernels ----------------
__global__ void dec_init(const float* __restrict__ v1, const float* __restrict__ v2,
                         const int* __restrict__ mask, float* __restrict__ prev,
                         float* __restrict__ first, int* __restrict__ mask_dec,
                         float* __restrict__ lps)
{
    int i = blockIdx.x * 256 + threadIdx.x;        // covers N*D
    if (i < N_ * D_) { prev[i] = v2[i & 255]; first[i] = v1[i & 255]; }
    if (i < N_ * S_) mask_dec[i] = (mask[i] == 0) ? 1 : 0;
    if (i < N_)      lps[i] = 0.0f;
}

__global__ void build_ctx(const float* __restrict__ g, const float* __restrict__ prev,
                          const float* __restrict__ first, float* __restrict__ ctx)
{
    int i = blockIdx.x * 256 + threadIdx.x;        // N*3D
    int n = i / (3 * D_), c = i % (3 * D_);
    float v = (c < D_)     ? g[n * D_ + c]
            : (c < 2 * D_) ? prev[n * D_ + (c - D_)]
                           : first[n * D_ + (c - 2 * D_)];
    ctx[i] = v;
}

// glance attention over S per batch element; one block per n, 256 threads
__global__ void __launch_bounds__(256)
dec_attn(const float* __restrict__ qg, const float* __restrict__ gk,
         const float* __restrict__ gv, const int* __restrict__ mask_dec,
         float* __restrict__ gl)
{
    __shared__ float qs[D_];
    __shared__ float sc[H_ * S_];
    const int n = blockIdx.x;
    const int tid = threadIdx.x;
    qs[tid] = qg[n * D_ + tid];
    __syncthreads();

    const float scale = 0.17677669529663687f;
    for (int p = tid; p < H_ * S_; p += 256) {
        int hh = p >> 7, s = p & 127;
        const float* kp = gk + ((size_t)s * N_ + n) * D_ + hh * DH_;
        float dot = 0.0f;
        #pragma unroll
        for (int d = 0; d < DH_; ++d) dot += qs[hh * DH_ + d] * kp[d];
        sc[p] = mask_dec[n * S_ + s] ? -1e9f : dot * scale;
    }
    __syncthreads();
    if (tid < H_) {
        float mx = -3.4e38f;
        for (int s = 0; s < S_; ++s) mx = fmaxf(mx, sc[tid * S_ + s]);
        float sum = 0.0f;
        for (int s = 0; s < S_; ++s) {
            float e = __expf(sc[tid * S_ + s] - mx);
            sum += e; sc[tid * S_ + s] = e;
        }
        float inv = 1.0f / sum;
        for (int s = 0; s < S_; ++s) sc[tid * S_ + s] *= inv;
    }
    __syncthreads();
    int hh = tid >> 5, dd = tid & 31;
    float out = 0.0f;
    for (int s = 0; s < S_; ++s)
        out += sc[hh * S_ + s] * gv[((size_t)s * N_ + n) * D_ + hh * DH_ + dd];
    gl[n * D_ + tid] = out;
}

// pointer logits + log-softmax + greedy select + state update
__global__ void __launch_bounds__(128)
dec_step(const float* __restrict__ ql, const float* __restrict__ pk,
         const float* __restrict__ h, int* __restrict__ mask_dec,
         float* __restrict__ prev, float* __restrict__ first,
         float* __restrict__ lps, int step)
{
    __shared__ float qs[D_];
    __shared__ float lg[S_];
    __shared__ float redv[S_];
    __shared__ int   redi[S_];
    const int n = blockIdx.x;
    const int s = threadIdx.x;

    qs[s] = ql[n * D_ + s];
    qs[s + 128] = ql[n * D_ + s + 128];
    __syncthreads();

    const float* pkp = pk + ((size_t)s * N_ + n) * D_;
    float dot = 0.0f;
    for (int d = 0; d < D_; ++d) dot += qs[d] * pkp[d];
    float logit = 10.0f * tanhf(dot);
    if (mask_dec[n * S_ + s]) logit = -1e9f;
    lg[s] = logit;

    // argmax (first max wins)
    redv[s] = logit; redi[s] = s; __syncthreads();
    for (int off = 64; off > 0; off >>= 1) {
        if (s < off) {
            float ov = redv[s + off]; int oi = redi[s + off];
            if (ov > redv[s] || (ov == redv[s] && oi < redi[s])) {
                redv[s] = ov; redi[s] = oi;
            }
        }
        __syncthreads();
    }
    const int idx = redi[0];
    const float mx = redv[0];
    __syncthreads();

    redv[s] = __expf(lg[s] - mx); __syncthreads();
    for (int off = 64; off > 0; off >>= 1) {
        if (s < off) redv[s] += redv[s + off];
        __syncthreads();
    }
    float lse = mx + __logf(redv[0]);

    if (s == 0) {
        lps[n] += lg[idx] - lse;
        mask_dec[n * S_ + idx] = 1;
    }
    const float* sel = h + ((size_t)idx * N_ + n) * D_;
    prev[n * D_ + s]       = sel[s];
    prev[n * D_ + s + 128] = sel[s + 128];
    if (step == 0) {
        first[n * D_ + s]       = sel[s];
        first[n * D_ + s + 128] = sel[s + 128];
    }
}

// ============================================================
// host launcher
// ============================================================
static inline size_t align256(size_t x) { return (x + 255) & ~(size_t)255; }

extern "C" void kernel_launch(void* const* d_in, const int* in_sizes, int n_in,
                              void* d_out, int out_size, void* d_ws, size_t ws_size,
                              hipStream_t stream)
{
    (void)in_sizes; (void)n_in; (void)out_size; (void)ws_size;

    const float* x        = (const float*)d_in[0];
    const int*   mask     = (const int*)  d_in[1];
    const float* qkv_w    = (const float*)d_in[2];
    const float* qkv_b    = (const float*)d_in[3];
    const float* out_w    = (const float*)d_in[4];
    const float* out_b    = (const float*)d_in[5];
    const float* ff1_w    = (const float*)d_in[6];
    const float* ff1_b    = (const float*)d_in[7];
    const float* ff2_w    = (const float*)d_in[8];
    const float* ff2_b    = (const float*)d_in[9];
    const float* ln1_s    = (const float*)d_in[10];
    const float* ln1_b    = (const float*)d_in[11];
    const float* ln2_s    = (const float*)d_in[12];
    const float* ln2_b    = (const float*)d_in[13];
    const float* v1       = (const float*)d_in[14];
    const float* v2       = (const float*)d_in[15];
    const float* g_wq     = (const float*)d_in[16];
    const float* g_wk     = (const float*)d_in[17];
    const float* g_wv     = (const float*)d_in[18];
    const float* g_wo     = (const float*)d_in[19];
    const float* g_bo     = (const float*)d_in[20];
    const float* p_wq     = (const float*)d_in[21];
    const float* p_wk     = (const float*)d_in[22];
    float* lps = (float*)d_out;

    // ---- workspace carve-up ----
    char* ws = (char*)d_ws;
    size_t off = 0;
    auto carve = [&](size_t bytes) { char* p = ws + off; off += align256(bytes); return p; };

    float* h       = (float*)carve((size_t)T_ * D_  * 4);         // 33.5 MB
    float* regionA = (float*)carve((size_t)T_ * FF_ * 4);         // 134 MB (qkv/ff1 -> gk,gv,pk)
    float* tmpO    = (float*)carve((size_t)T_ * D_  * 4);         // attention out
    float* tmp2    = (float*)carve((size_t)T_ * D_  * 4);         // proj / ff2 out
    float* graph   = (float*)carve((size_t)N_ * D_ * 4);
    float* ctx     = (float*)carve((size_t)N_ * 3 * D_ * 4);
    float* qg      = (float*)carve((size_t)N_ * D_ * 4);
    float* gl      = (float*)carve((size_t)N_ * D_ * 4);
    float* glo     = (float*)carve((size_t)N_ * D_ * 4);
    float* qlb     = (float*)carve((size_t)N_ * D_ * 4);
    float* prev    = (float*)carve((size_t)N_ * D_ * 4);
    float* first   = (float*)carve((size_t)N_ * D_ * 4);
    int*   mask_dec= (int*)  carve((size_t)N_ * S_ * 4);

    unsigned short* qkv_wb = (unsigned short*)carve((size_t)NL_ * D_ * 3 * D_ * 2);
    unsigned short* out_wb = (unsigned short*)carve((size_t)NL_ * D_ * D_ * 2);
    unsigned short* ff1_wb = (unsigned short*)carve((size_t)NL_ * D_ * FF_ * 2);
    unsigned short* ff2_wb = (unsigned short*)carve((size_t)NL_ * FF_ * D_ * 2);
    unsigned short* gwq_b  = (unsigned short*)carve((size_t)3 * D_ * D_ * 2);
    unsigned short* gwk_b  = (unsigned short*)carve((size_t)D_ * D_ * 2);
    unsigned short* gwv_b  = (unsigned short*)carve((size_t)D_ * D_ * 2);
    unsigned short* gwo_b  = (unsigned short*)carve((size_t)D_ * D_ * 2);
    unsigned short* pwq_b  = (unsigned short*)carve((size_t)D_ * D_ * 2);
    unsigned short* pwk_b  = (unsigned short*)carve((size_t)D_ * D_ * 2);

    // decode K/V/pointer-key caches alias regionA after the encoder
    float* gk = regionA;
    float* gv = regionA + (size_t)T_ * D_;
    float* pk = regionA + (size_t)2 * T_ * D_;

    // ---- weight conversion (fp32 -> bf16, fragment-permuted) ----
    auto convw = [&](const float* src, unsigned short* dst, int K, int Nc) {
        int cnt = K * Nc;
        f32_to_bf16_wperm<<<(cnt + 255) / 256, 256, 0, stream>>>(src, dst, K, Nc);
    };
    for (int l = 0; l < NL_; ++l) {
        convw(qkv_w + (size_t)l * D_ * 3 * D_, qkv_wb + (size_t)l * D_ * 3 * D_, D_, 3 * D_);
        convw(out_w + (size_t)l * D_ * D_,     out_wb + (size_t)l * D_ * D_,     D_, D_);
        convw(ff1_w + (size_t)l * D_ * FF_,    ff1_wb + (size_t)l * D_ * FF_,    D_, FF_);
        convw(ff2_w + (size_t)l * FF_ * D_,    ff2_wb + (size_t)l * FF_ * D_,    FF_, D_);
    }
    convw(g_wq, gwq_b, 3 * D_, D_);
    convw(g_wk, gwk_b, D_, D_);
    convw(g_wv, gwv_b, D_, D_);
    convw(g_wo, gwo_b, D_, D_);
    convw(p_wq, pwq_b, D_, D_);
    convw(p_wk, pwk_b, D_, D_);

    // h = x
    hipMemcpyAsync(h, x, (size_t)T_ * D_ * 4, hipMemcpyDeviceToDevice, stream);

    auto gemm = [&](const float* A, const unsigned short* W, const float* bias,
                    float* C, int M, int K, int Nc, float alpha, int relu) {
        dim3 grid(Nc / 64, M / 128);
        gemm_wmma<<<grid, 256, 0, stream>>>(A, W, bias, C, M, K, Nc, alpha, relu);
    };

    const size_t attn_lds = (size_t)3 * 128 * 32 * 2          // Q,K,V bf16
                          + (size_t)128 * SC_STRIDE * 4       // scores f32 (padded)
                          + (size_t)128 * PS_STRIDE * 2;      // P bf16 (padded)

    // ---- encoder ----
    for (int l = 0; l < NL_; ++l) {
        gemm(h, qkv_wb + (size_t)l * D_ * 3 * D_, qkv_b + l * 3 * D_,
             regionA, T_, D_, 3 * D_, 1.0f, 0);
        enc_attn<<<N_ * H_, 256, attn_lds, stream>>>(regionA, mask, tmpO);
        gemm(tmpO, out_wb + (size_t)l * D_ * D_, out_b + l * D_,
             tmp2, T_, D_, D_, 1.0f, 0);
        add_ln<<<T_, 256, 0, stream>>>(h, tmp2, ln1_s + l * D_, ln1_b + l * D_);
        gemm(h, ff1_wb + (size_t)l * D_ * FF_, ff1_b + l * FF_,
             regionA, T_, D_, FF_, 1.0f, 1);
        gemm(regionA, ff2_wb + (size_t)l * FF_ * D_, ff2_b + l * D_,
             tmp2, T_, FF_, D_, 1.0f, 0);
        add_ln<<<T_, 256, 0, stream>>>(h, tmp2, ln2_s + l * D_, ln2_b + l * D_);
    }

    // ---- decode-side precompute ----
    graph_mean<<<(N_ * D_) / 256, 256, 0, stream>>>(h, graph);
    gemm(h, gwk_b, nullptr, gk, T_, D_, D_, 1.0f, 0);
    gemm(h, gwv_b, nullptr, gv, T_, D_, D_, 1.0f, 0);
    gemm(h, pwk_b, nullptr, pk, T_, D_, D_, 1.0f, 0);

    dec_init<<<(N_ * D_) / 256, 256, 0, stream>>>(v1, v2, mask, prev, first, mask_dec, lps);

    // ---- sequential greedy decode (128 steps, graph-captured) ----
    const float inv_sqrtD = 1.0f / 16.0f;  // 1/sqrt(256)
    for (int t = 0; t < S_; ++t) {
        build_ctx<<<(N_ * 3 * D_) / 256, 256, 0, stream>>>(graph, prev, first, ctx);
        gemm(ctx, gwq_b, nullptr, qg, N_, 3 * D_, D_, 1.0f, 0);
        dec_attn<<<N_, 256, 0, stream>>>(qg, gk, gv, mask_dec, gl);
        gemm(gl, gwo_b, g_bo, glo, N_, D_, D_, 1.0f, 0);
        gemm(glo, pwq_b, nullptr, qlb, N_, D_, D_, inv_sqrtD, 0);
        dec_step<<<N_, 128, 0, stream>>>(qlb, pk, h, mask_dec, prev, first, lps, t);
    }
}
